// Qwen3MoeDecoderLayer_88235808129238
// MI455X (gfx1250) — compile-verified, matrix operand order checked by reading, not run on hardware
//
#include <hip/hip_runtime.h>
#include <hip/hip_bf16.h>

#define T_    1024
#define H_    2048
#define NH_   32
#define NKV_  4
#define HD_   128
#define DQ_   (NH_*HD_)     // 4096
#define DKV_  (NKV_*HD_)    // 512
#define DQKV_ (DQ_+2*DKV_)  // 5120
#define E_    32
#define I_    768
#define TOPK_ 8
#define EPS_  1e-6f
#define KCHUNK_ 2048

typedef __bf16 bf16;
typedef bf16  v16bf __attribute__((ext_vector_type(16)));
typedef bf16  v8bf  __attribute__((ext_vector_type(8)));
typedef float v8f   __attribute__((ext_vector_type(8)));
typedef unsigned int v4u __attribute__((ext_vector_type(4)));
typedef int   v8i   __attribute__((ext_vector_type(8)));
typedef int   v4i   __attribute__((ext_vector_type(4)));

__device__ __forceinline__ v8f v8f_zero() {
    v8f z;
#pragma unroll
    for (int i = 0; i < 8; ++i) z[i] = 0.0f;
    return z;
}

__device__ __forceinline__ v8f wmma_bf16(v16bf a, v16bf b, v8f c) {
    return __builtin_amdgcn_wmma_f32_16x16x32_bf16(false, a, false, b, (short)0, c,
                                                   false, false);
}

// A/B fragment from a row-major bf16 row: elements 0..7 at p[0..7], 8..15 at p[16..23]
__device__ __forceinline__ v16bf frag_from_rows(const bf16* p) {
    v8bf lo = *(const v8bf*)p;
    v8bf hi = *(const v8bf*)(p + 16);
    v16bf r;
#pragma unroll
    for (int i = 0; i < 8; ++i) { r[i] = lo[i]; r[i + 8] = hi[i]; }
    return r;
}

// Same but source is fp32 in memory, converted to bf16 (RNE) in registers
__device__ __forceinline__ v16bf frag_from_f32(const float* p) {
    v8f lo = *(const v8f*)p;
    v8f hi = *(const v8f*)(p + 16);
    v16bf r;
#pragma unroll
    for (int i = 0; i < 8; ++i) { r[i] = (bf16)lo[i]; r[i + 8] = (bf16)hi[i]; }
    return r;
}

// B fragment gathered column-wise (per-lane column, k-strided rows)
__device__ __forceinline__ v16bf frag_gather_col(const bf16* base, int stride) {
    v16bf r;
#pragma unroll
    for (int e = 0; e < 16; ++e) {
        int k = (e & 7) + ((e >> 3) << 4);
        r[e] = base[(size_t)k * stride];
    }
    return r;
}

// ---- Tensor Data Mover: issue one 2D tile load (bf16 elements) into LDS offset 0 ----
__device__ __forceinline__ void tdm_issue(v4u g0, v8i g1, v4i g2, v4i g3) {
#if defined(__clang_major__) && __clang_major__ >= 23
    v8i z8 = {0, 0, 0, 0, 0, 0, 0, 0};
    __builtin_amdgcn_tensor_load_to_lds(g0, g1, g2, g3, z8, 0);
#else
    __builtin_amdgcn_tensor_load_to_lds(g0, g1, g2, g3, 0);
#endif
}

__device__ __forceinline__ void tdm_build_g01(v4u& g0, v8i& g1, unsigned lds_addr,
                                              const void* gptr, unsigned tdim0,
                                              unsigned tdim1, unsigned tile0,
                                              unsigned tile1, unsigned stride0) {
    unsigned long long ga = (unsigned long long)gptr;
    g0[0] = 1u;                                   // count=1 valid user descriptor
    g0[1] = lds_addr;                             // LDS byte address
    g0[2] = (unsigned)ga;                         // global_addr[31:0]
    g0[3] = (unsigned)((ga >> 32) & 0x1FFFFFFu) | (2u << 30);   // addr[56:32] | type=2
    g1[0] = (int)(1u << 16);                      // data_size = 2 bytes
    g1[1] = (int)((tdim0 & 0xFFFFu) << 16);       // tensor_dim0[15:0]
    g1[2] = (int)(((tdim0 >> 16) & 0xFFFFu) | ((tdim1 & 0xFFFFu) << 16));
    g1[3] = (int)(((tdim1 >> 16) & 0xFFFFu) | ((tile0 & 0xFFFFu) << 16));
    g1[4] = (int)(tile1 & 0xFFFFu);               // tile_dim1 (tile_dim2 = 0)
    g1[5] = (int)stride0;                         // tensor_dim0_stride[31:0]
    g1[6] = 0;                                    // stride0[47:32] | dim1_stride lo
    g1[7] = 0;
}

// Plain 2D tile: tdim1 rows valid (OOB rows read as zero), tile1 rows transferred
__device__ __forceinline__ void tdm_load_2d(unsigned lds_addr, const void* gptr,
                                            unsigned tdim0, unsigned tdim1,
                                            unsigned tile0, unsigned tile1,
                                            unsigned stride0) {
    v4u g0; v8i g1;
    tdm_build_g01(g0, g1, lds_addr, gptr, tdim0, tdim1, tile0, tile1, stride0);
    v4i z = {0, 0, 0, 0};
    tdm_issue(g0, g1, z, z);
}

// Gather mode: up to 16 rows selected by 16-bit row indices (ascending)
__device__ __forceinline__ void tdm_load_gather16(unsigned lds_addr, const void* gptr,
                                                  unsigned tdim0, unsigned tile0,
                                                  unsigned nvalid, unsigned stride0,
                                                  const unsigned* idx) {
    v4u g0; v8i g1;
    tdm_build_g01(g0, g1, lds_addr, gptr, tdim0, /*tdim1*/ T_, tile0, nvalid, stride0);
    g0[0] |= (1u << 31);                          // gather_mode, 16-bit indices
    v4i g2, g3;
#pragma unroll
    for (int i = 0; i < 4; ++i) {
        g2[i] = (int)((idx[2 * i] & 0xFFFFu) | ((idx[2 * i + 1] & 0xFFFFu) << 16));
        g3[i] = (int)((idx[8 + 2 * i] & 0xFFFFu) | ((idx[8 + 2 * i + 1] & 0xFFFFu) << 16));
    }
    tdm_issue(g0, g1, g2, g3);
}

// ---------------- RMSNorm (row per block) ----------------
__global__ __launch_bounds__(256) void k_rmsnorm(const float* __restrict__ x,
                                                 const float* __restrict__ w,
                                                 bf16* __restrict__ ob,
                                                 float* __restrict__ of, int D) {
    int t = blockIdx.x;
    const float* xr = x + (size_t)t * D;
    __shared__ float sh[8];
    int tid = threadIdx.x, lane = tid & 31, wv = tid >> 5;
    float s = 0.f;
    for (int i = tid; i < D; i += 256) { float v = xr[i]; s += v * v; }
#pragma unroll
    for (int m = 1; m < 32; m <<= 1) s += __shfl_xor(s, m, 32);
    if (lane == 0) sh[wv] = s;
    __syncthreads();
    if (tid == 0) {
        float a = 0.f;
#pragma unroll
        for (int i = 0; i < 8; ++i) a += sh[i];
        sh[0] = rsqrtf(a / (float)D + EPS_);
    }
    __syncthreads();
    float sc = sh[0];
    for (int i = tid; i < D; i += 256) {
        float v = xr[i] * sc * w[i];
        ob[(size_t)t * D + i] = (bf16)v;
        if (of) of[(size_t)t * D + i] = v;
    }
}

// ------- Generic WMMA GEMM with TDM-staged A tile in LDS -------
// C[T x N] = A[T x Ka](bf16) * W[N x Ka](f32)^T  (+ optional residual)
__global__ __launch_bounds__(256) void k_gemm(const bf16* __restrict__ A,
                                              const float* __restrict__ W,
                                              const float* __restrict__ addsrc,
                                              float* __restrict__ Cout,
                                              int Ka, int N) {
    extern __shared__ __align__(16) char dsm[];
    const bf16* As = (const bf16*)dsm;
    int lane = threadIdx.x & 31, wid = threadIdx.x >> 5;
    int n0 = blockIdx.x * 128 + wid * 16;
    int m0 = blockIdx.y * 16;
    int l15 = lane & 15, g = lane >> 4;
    const float* Wp = W + (size_t)(n0 + l15) * Ka + g * 8;
    v8f c = v8f_zero();
    for (int kc = 0; kc < Ka; kc += KCHUNK_) {
        int kl = (Ka - kc) < KCHUNK_ ? (Ka - kc) : KCHUNK_;
        __syncthreads();                 // prior chunk fully consumed
        if (wid == 0) {
            tdm_load_2d(0u, A + (size_t)m0 * Ka + kc,
                        (unsigned)kl, 16u, (unsigned)kl, 16u, (unsigned)Ka);
            __builtin_amdgcn_s_wait_tensorcnt(0);
        }
        __syncthreads();
        const bf16* Ap = As + (size_t)l15 * kl + g * 8;
        const float* Wk = Wp + kc;
        for (int k0 = 0; k0 < kl; k0 += 32) {
            __builtin_prefetch(Wk + k0 + 256, 0, 1);   // stream weights ~1KB ahead
            v16bf a = frag_from_rows(Ap + k0);
            v16bf b = frag_from_f32(Wk + k0);
            c = wmma_bf16(a, b, c);
        }
    }
    int n = l15, mo = g * 8;
#pragma unroll
    for (int r = 0; r < 8; ++r) {
        size_t idx = (size_t)(m0 + mo + r) * N + n0 + n;
        float v = c[r];
        if (addsrc) v += addsrc[idx];
        Cout[idx] = v;
    }
}

// ---------------- Per-head q/k RMSNorm + RoPE, v convert ----------------
__global__ __launch_bounds__(128) void k_qknorm_rope(const float* __restrict__ qkv,
                                                     const int* __restrict__ positions,
                                                     const float* __restrict__ qw,
                                                     const float* __restrict__ kw,
                                                     bf16* __restrict__ qb,
                                                     bf16* __restrict__ kb,
                                                     bf16* __restrict__ vb) {
    int t = blockIdx.x, d = threadIdx.x;   // 128 threads
    float pos = (float)positions[t];
    __shared__ float xn[HD_];
    __shared__ float red[4];
    int lane = d & 31, wv = d >> 5;
    for (int hh = 0; hh < NH_ + NKV_; ++hh) {
        const float* src; const float* wgt; bf16* dst;
        if (hh < NH_) {
            src = qkv + (size_t)t * DQKV_ + hh * HD_;
            wgt = qw;  dst = qb + (size_t)t * DQ_ + hh * HD_;
        } else {
            int kh = hh - NH_;
            src = qkv + (size_t)t * DQKV_ + DQ_ + kh * HD_;
            wgt = kw;  dst = kb + (size_t)t * DKV_ + kh * HD_;
        }
        float v = src[d];
        float s = v * v;
#pragma unroll
        for (int m = 1; m < 32; m <<= 1) s += __shfl_xor(s, m, 32);
        if (lane == 0) red[wv] = s;
        __syncthreads();
        float tot = red[0] + red[1] + red[2] + red[3];
        float sc = rsqrtf(tot / (float)HD_ + EPS_);
        xn[d] = v * sc * wgt[d];
        __syncthreads();
        if (d < 64) {
            float inv = __powf(10000.0f, -(float)d / 64.0f);
            float ang = pos * inv;
            float c, si;
            __sincosf(ang, &c, &si);
            float x1 = xn[d], x2 = xn[d + 64];
            dst[d]      = (bf16)(x1 * c - x2 * si);
            dst[d + 64] = (bf16)(x2 * c + x1 * si);
        }
        __syncthreads();
    }
    for (int i = d; i < DKV_; i += 128)
        vb[(size_t)t * DKV_ + i] = (bf16)qkv[(size_t)t * DQKV_ + DQ_ + DKV_ + i];
}

// ---------------- Flash attention: one wave per (head, 16-query tile) -------------
__global__ __launch_bounds__(256) void k_attn(const bf16* __restrict__ qb,
                                              const bf16* __restrict__ kb,
                                              const bf16* __restrict__ vb,
                                              bf16* __restrict__ ob) {
    __shared__ __align__(16) bf16 plds[8][16 * 32];
    int lane = threadIdx.x & 31, wid = threadIdx.x >> 5;
    int wg = blockIdx.x * 8 + wid;
    int head = wg >> 6;          // 64 q-tiles per head
    int qt = wg & 63;
    int q0 = qt * 16;
    int kvh = head >> 3;         // NH/NKV = 8
    const float scale = 0.08838834764831845f;   // 1/sqrt(128)
    int mlan = lane & 15, g = lane >> 4, mo = g * 8;

    const bf16* Qp = qb + (size_t)(q0 + mlan) * DQ_ + head * HD_ + g * 8;
    v16bf aq[4];
#pragma unroll
    for (int c = 0; c < 4; ++c) aq[c] = frag_from_rows(Qp + c * 32);

    v8f oacc[8];
#pragma unroll
    for (int f = 0; f < 8; ++f) oacc[f] = v8f_zero();
    float mr[8], lr[8];
#pragma unroll
    for (int r = 0; r < 8; ++r) { mr[r] = -3.0e38f; lr[r] = 0.f; }

    bf16* myp = &plds[wid][0];

    for (int kbs = 0; kbs <= q0 + 15; kbs += 32) {
        // ---- S = Q K^T (16x32 as two 16x16 subtiles) ----
        v8f s0 = v8f_zero(), s1 = v8f_zero();
        const bf16* Kp0 = kb + (size_t)(kbs + mlan) * DKV_ + kvh * HD_ + g * 8;
        const bf16* Kp1 = kb + (size_t)(kbs + 16 + mlan) * DKV_ + kvh * HD_ + g * 8;
#pragma unroll
        for (int c = 0; c < 4; ++c) {
            s0 = wmma_bf16(aq[c], frag_from_rows(Kp0 + c * 32), s0);
            s1 = wmma_bf16(aq[c], frag_from_rows(Kp1 + c * 32), s1);
        }
        // ---- online softmax ----
        float alpha[8];
#pragma unroll
        for (int r = 0; r < 8; ++r) {
            int mq = q0 + mo + r;
            int j0 = kbs + mlan, j1 = kbs + 16 + mlan;
            float v0 = s0[r] * scale; if (j0 > mq) v0 = -3.0e38f;
            float v1 = s1[r] * scale; if (j1 > mq) v1 = -3.0e38f;
            float mx = fmaxf(v0, v1);
#pragma unroll
            for (int m = 1; m < 16; m <<= 1) mx = fmaxf(mx, __shfl_xor(mx, m, 32));
            float mnew = fmaxf(mr[r], mx);
            float p0 = __expf(v0 - mnew), p1 = __expf(v1 - mnew);
            float rs = p0 + p1;
#pragma unroll
            for (int m = 1; m < 16; m <<= 1) rs += __shfl_xor(rs, m, 32);
            alpha[r] = __expf(mr[r] - mnew);
            lr[r] = lr[r] * alpha[r] + rs;
            mr[r] = mnew;
            myp[(mo + r) * 32 + mlan]      = (bf16)p0;
            myp[(mo + r) * 32 + 16 + mlan] = (bf16)p1;
        }
#pragma unroll
        for (int f = 0; f < 8; ++f)
#pragma unroll
            for (int r = 0; r < 8; ++r) oacc[f][r] *= alpha[r];
        asm volatile("s_wait_dscnt 0x0" ::: "memory");
        // ---- P (A-layout) from LDS, then O += P V ----
        v16bf ap = frag_from_rows(myp + mlan * 32 + g * 8);
#pragma unroll
        for (int f = 0; f < 8; ++f) {
            const bf16* Vp = vb + (size_t)(kbs + g * 8) * DKV_ + kvh * HD_ + f * 16 + mlan;
            oacc[f] = wmma_bf16(ap, frag_gather_col(Vp, DKV_), oacc[f]);
        }
    }
    // ---- epilogue ----
#pragma unroll
    for (int f = 0; f < 8; ++f)
#pragma unroll
        for (int r = 0; r < 8; ++r) {
            float v = oacc[f][r] / lr[r];
            ob[(size_t)(q0 + mo + r) * DQ_ + head * HD_ + f * 16 + mlan] = (bf16)v;
        }
}

// ---------------- Router: logits, softmax, top-8, normalize ----------------
__global__ __launch_bounds__(256) void k_router(const float* __restrict__ h,
                                                const float* __restrict__ gate_w,
                                                int* __restrict__ topk_idx,
                                                float* __restrict__ topk_w) {
    int t = blockIdx.x;
    __shared__ float logits[E_];
    int gi = threadIdx.x >> 3, li = threadIdx.x & 7;   // 32 groups of 8 lanes
    const float* hr = h + (size_t)t * H_;
    const float* wr = gate_w + (size_t)gi * H_;
    float s = 0.f;
    for (int i = li; i < H_; i += 8) s += hr[i] * wr[i];
#pragma unroll
    for (int m = 1; m < 8; m <<= 1) s += __shfl_xor(s, m, 32);
    if (li == 0) logits[gi] = s;
    __syncthreads();
    if (threadIdx.x == 0) {
        float mx = -3.0e38f;
        for (int e = 0; e < E_; ++e) mx = fmaxf(mx, logits[e]);
        float p[E_]; float den = 0.f;
        for (int e = 0; e < E_; ++e) { p[e] = __expf(logits[e] - mx); den += p[e]; }
        float inv = 1.0f / den;
        for (int e = 0; e < E_; ++e) p[e] *= inv;
        float wsum = 0.f;
        int idx[TOPK_]; float val[TOPK_];
        for (int k = 0; k < TOPK_; ++k) {
            float best = -1.f; int bi = 0;
            for (int e = 0; e < E_; ++e) if (p[e] > best) { best = p[e]; bi = e; }
            idx[k] = bi; val[k] = best; p[bi] = -2.f; wsum += best;
        }
        float iw = 1.0f / wsum;
        for (int k = 0; k < TOPK_; ++k) {
            topk_idx[t * TOPK_ + k] = idx[k];
            topk_w[t * TOPK_ + k] = val[k] * iw;
        }
    }
}

// ---------------- Deterministic expert bucketing ----------------
__global__ __launch_bounds__(256) void k_count(const int* __restrict__ topk_idx,
                                               int* __restrict__ counts) {
    int e = blockIdx.x, tid = threadIdx.x;
    __shared__ int sh[256];
    int c = 0, t0 = tid * 4;
    for (int t = t0; t < t0 + 4; ++t)
        for (int s = 0; s < TOPK_; ++s) if (topk_idx[t * TOPK_ + s] == e) c++;
    sh[tid] = c;
    __syncthreads();
    for (int o = 128; o > 0; o >>= 1) { if (tid < o) sh[tid] += sh[tid + o]; __syncthreads(); }
    if (tid == 0) counts[e] = sh[0];
}

__global__ void k_offsets(const int* __restrict__ counts, int* __restrict__ offsets) {
    if (threadIdx.x == 0) {
        int a = 0;
        for (int e = 0; e < E_; ++e) { offsets[e] = a; a += counts[e]; }
        offsets[E_] = a;
    }
}

__global__ __launch_bounds__(256) void k_scatter(const int* __restrict__ topk_idx,
                                                 const float* __restrict__ topk_wv,
                                                 const int* __restrict__ offsets,
                                                 int* __restrict__ slot_tok,
                                                 float* __restrict__ slot_w) {
    int e = blockIdx.x, tid = threadIdx.x;
    __shared__ int sh[256];
    int c = 0, t0 = tid * 4;
    for (int t = t0; t < t0 + 4; ++t)
        for (int s = 0; s < TOPK_; ++s) if (topk_idx[t * TOPK_ + s] == e) c++;
    sh[tid] = c;
    __syncthreads();
    if (tid == 0) { int a = 0; for (int i = 0; i < 256; ++i) { int v = sh[i]; sh[i] = a; a += v; } }
    __syncthreads();
    int pos = offsets[e] + sh[tid];
    for (int t = t0; t < t0 + 4; ++t)
        for (int s = 0; s < TOPK_; ++s)
            if (topk_idx[t * TOPK_ + s] == e) {
                slot_tok[pos] = t;
                slot_w[pos] = topk_wv[t * TOPK_ + s];
                pos++;
            }
}

__global__ void k_copy(const float* __restrict__ src, float* __restrict__ dst, int n) {
    int i = blockIdx.x * 256 + threadIdx.x;
    if (i < n) dst[i] = src[i];
}

// --------- MoE GEMM1: act = silu(h Wg^T) * (h Wu^T), A rows via TDM gather ---------
__global__ __launch_bounds__(256) void k_moe1(const bf16* __restrict__ hpost,
                                              const float* __restrict__ w_gate,
                                              const float* __restrict__ w_up,
                                              const int* __restrict__ counts,
                                              const int* __restrict__ offsets,
                                              const int* __restrict__ slot_tok,
                                              bf16* __restrict__ act) {
    extern __shared__ __align__(16) char dsm[];
    const bf16* As = (const bf16*)dsm;
    int e = blockIdx.z;
    int ne = counts[e];
    int row_base = blockIdx.y * 16;
    if (row_base >= ne) return;
    int lane = threadIdx.x & 31, wid = threadIdx.x >> 5;
    int n0 = blockIdx.x * 128 + wid * 16;
    int l15 = lane & 15, g = lane >> 4;
    int base = offsets[e];
    if (wid == 0) {
        unsigned idx[16];
#pragma unroll
        for (int r = 0; r < 16; ++r) {
            int rr = row_base + r; if (rr > ne - 1) rr = ne - 1;
            idx[r] = (unsigned)slot_tok[base + rr];     // ascending token ids
        }
        unsigned nval = (unsigned)(ne - row_base); if (nval > 16u) nval = 16u;
        tdm_load_gather16(0u, hpost, (unsigned)H_, (unsigned)H_, nval, (unsigned)H_, idx);
        __builtin_amdgcn_s_wait_tensorcnt(0);
    }
    __syncthreads();
    const bf16*  Ap = As + (size_t)l15 * H_ + g * 8;
    const float* Gp = w_gate + ((size_t)e * I_ + n0 + l15) * H_ + g * 8;
    const float* Up = w_up   + ((size_t)e * I_ + n0 + l15) * H_ + g * 8;
    v8f cg = v8f_zero(), cu = v8f_zero();
    for (int k0 = 0; k0 < H_; k0 += 32) {
        __builtin_prefetch(Gp + k0 + 256, 0, 1);
        __builtin_prefetch(Up + k0 + 256, 0, 1);
        v16bf a = frag_from_rows(Ap + k0);
        cg = wmma_bf16(a, frag_from_f32(Gp + k0), cg);
        cu = wmma_bf16(a, frag_from_f32(Up + k0), cu);
    }
    int n = l15, mo = g * 8;
#pragma unroll
    for (int r = 0; r < 8; ++r) {
        int row = row_base + mo + r;
        if (row < ne) {
            float gv = cg[r], uv = cu[r];
            float sv = gv / (1.0f + __expf(-gv)) * uv;
            act[(size_t)(base + row) * I_ + n0 + n] = (bf16)sv;
        }
    }
}

// --------- MoE GEMM2: out += cw * (act Wd^T), A tile via 2D TDM (OOB rows -> 0) -----
__global__ __launch_bounds__(256) void k_moe2(const bf16* __restrict__ act,
                                              const float* __restrict__ w_down,
                                              const int* __restrict__ counts,
                                              const int* __restrict__ offsets,
                                              const int* __restrict__ slot_tok,
                                              const float* __restrict__ slot_w,
                                              float* __restrict__ out) {
    extern __shared__ __align__(16) char dsm[];
    const bf16* As = (const bf16*)dsm;
    int e = blockIdx.z;
    int ne = counts[e];
    int row_base = blockIdx.y * 16;
    if (row_base >= ne) return;
    int lane = threadIdx.x & 31, wid = threadIdx.x >> 5;
    int n0 = blockIdx.x * 128 + wid * 16;
    int l15 = lane & 15, g = lane >> 4;
    int base = offsets[e];
    if (wid == 0) {
        tdm_load_2d(0u, act + (size_t)(base + row_base) * I_,
                    (unsigned)I_, (unsigned)(ne - row_base), (unsigned)I_, 16u,
                    (unsigned)I_);
        __builtin_amdgcn_s_wait_tensorcnt(0);
    }
    __syncthreads();
    const bf16*  Ap = As + (size_t)l15 * I_ + g * 8;
    const float* Wp = w_down + ((size_t)e * H_ + n0 + l15) * I_ + g * 8;
    v8f c = v8f_zero();
    for (int k0 = 0; k0 < I_; k0 += 32) {
        __builtin_prefetch(Wp + k0 + 256, 0, 1);
        c = wmma_bf16(frag_from_rows(Ap + k0), frag_from_f32(Wp + k0), c);
    }
    int n = l15, mo = g * 8;
#pragma unroll
    for (int r = 0; r < 8; ++r) {
        int row = row_base + mo + r;
        if (row < ne) {
            int slot = base + row;
            int tok = slot_tok[slot];
            float wv = slot_w[slot];
            atomicAdd(&out[(size_t)tok * H_ + n0 + n], c[r] * wv);
        }
    }
}

// ---------------- Host launch ----------------
extern "C" void kernel_launch(void* const* d_in, const int* in_sizes, int n_in,
                              void* d_out, int out_size, void* d_ws, size_t ws_size,
                              hipStream_t stream) {
    const int*   positions = (const int*)d_in[0];
    const float* hidden    = (const float*)d_in[1];
    const float* qkv_w     = (const float*)d_in[2];
    const float* o_w       = (const float*)d_in[3];
    const float* q_norm_w  = (const float*)d_in[4];
    const float* k_norm_w  = (const float*)d_in[5];
    const float* in_ln_w   = (const float*)d_in[6];
    const float* post_ln_w = (const float*)d_in[7];
    const float* gate_w    = (const float*)d_in[8];
    const float* w_gate    = (const float*)d_in[9];
    const float* w_up      = (const float*)d_in[10];
    const float* w_down    = (const float*)d_in[11];
    float* out = (float*)d_out;

    char* w = (char*)d_ws;
    auto carve = [&](size_t bytes) {
        char* p = w;
        w += (bytes + 255) & ~(size_t)255;
        return p;
    };
    bf16*  h_ln      = (bf16*)carve((size_t)T_ * H_ * 2);
    float* qkv       = (float*)carve((size_t)T_ * DQKV_ * 4);
    bf16*  qb        = (bf16*)carve((size_t)T_ * DQ_ * 2);
    bf16*  kbuf      = (bf16*)carve((size_t)T_ * DKV_ * 2);
    bf16*  vbuf      = (bf16*)carve((size_t)T_ * DKV_ * 2);
    bf16*  attn      = (bf16*)carve((size_t)T_ * DQ_ * 2);
    float* resid2    = (float*)carve((size_t)T_ * H_ * 4);
    float* hpost_f   = (float*)carve((size_t)T_ * H_ * 4);
    bf16*  hpost_b   = (bf16*)carve((size_t)T_ * H_ * 2);
    int*   topk_idx  = (int*)carve((size_t)T_ * TOPK_ * 4);
    float* topk_w    = (float*)carve((size_t)T_ * TOPK_ * 4);
    int*   counts    = (int*)carve(E_ * 4);
    int*   offsets   = (int*)carve((E_ + 1) * 4);
    int*   slot_tok  = (int*)carve((size_t)T_ * TOPK_ * 4);
    float* slot_w    = (float*)carve((size_t)T_ * TOPK_ * 4);
    bf16*  actbuf    = (bf16*)carve((size_t)T_ * TOPK_ * I_ * 2);

    const size_t ldsA  = (size_t)16 * KCHUNK_ * 2;   // 64 KB A-tile stage
    const size_t ldsA2 = (size_t)16 * I_ * 2;        // 24 KB for moe2

    // 1. input RMSNorm -> bf16
    k_rmsnorm<<<T_, 256, 0, stream>>>(hidden, in_ln_w, h_ln, nullptr, H_);
    // 2. QKV projection (TDM-staged A)
    k_gemm<<<dim3(DQKV_ / 128, T_ / 16), 256, ldsA, stream>>>(h_ln, qkv_w, nullptr, qkv,
                                                              H_, DQKV_);
    // 3. q/k RMSNorm + RoPE, v->bf16
    k_qknorm_rope<<<T_, 128, 0, stream>>>(qkv, positions, q_norm_w, k_norm_w, qb, kbuf, vbuf);
    // 4. flash attention (2048 waves)
    k_attn<<<(NH_ * (T_ / 16)) / 8, 256, 0, stream>>>(qb, kbuf, vbuf, attn);
    // 5. O projection + residual (K=4096 -> two TDM chunks)
    k_gemm<<<dim3(H_ / 128, T_ / 16), 256, ldsA, stream>>>(attn, o_w, hidden, resid2,
                                                           DQ_, H_);
    // 6. post RMSNorm
    k_rmsnorm<<<T_, 256, 0, stream>>>(resid2, post_ln_w, hpost_b, hpost_f, H_);
    // 7. router + top-8
    k_router<<<T_, 256, 0, stream>>>(hpost_f, gate_w, topk_idx, topk_w);
    // 8-10. deterministic expert bucketing
    k_count<<<E_, 256, 0, stream>>>(topk_idx, counts);
    k_offsets<<<1, 32, 0, stream>>>(counts, offsets);
    k_scatter<<<E_, 256, 0, stream>>>(topk_idx, topk_w, offsets, slot_tok, slot_w);
    // 11. out = residual (moe accumulates on top)
    k_copy<<<(T_ * H_ + 255) / 256, 256, 0, stream>>>(resid2, out, T_ * H_);
    // 12. expert gate/up + SiLU (TDM gather-mode token rows)
    k_moe1<<<dim3(I_ / 128, T_ / 16, E_), 256, ldsA, stream>>>(hpost_b, w_gate, w_up,
                                                               counts, offsets, slot_tok,
                                                               actbuf);
    // 13. expert down + weighted accumulate (2D TDM, OOB rows zero-filled)
    k_moe2<<<dim3(H_ / 128, T_ / 16, E_), 256, ldsA2, stream>>>(actbuf, w_down, counts,
                                                                offsets, slot_tok, slot_w,
                                                                out);
    (void)in_sizes; (void)n_in; (void)out_size; (void)ws_size;
}
